// PVConv_23957327577806
// MI455X (gfx1250) — matrix operand order repaired and checked by reading, not compiled, not run on hardware
//
#include <hip/hip_runtime.h>
#include <hip/hip_bf16.h>
#include <stdint.h>

typedef _Float16 v16h __attribute__((ext_vector_type(16)));
typedef float    v8f  __attribute__((ext_vector_type(8)));

#define NPTS   32768
#define RCUBE  32768

// ---------------- workspace layout (bytes) ----------------
#define OFF_SUMS   0ull                       // 64MB f32 [B][64][32768]; reused: conv1_out, conv2_out, p_out
#define OFF_CNT    (OFF_SUMS  + 67108864ull)  // 1MB  f32 [B][32768]
#define OFF_G16    (OFF_CNT   + 1048576ull)   // 32MB f16 [B][64][32768]; reused as h_f16
#define OFF_FTST   (OFF_G16   + 33554432ull)  // 32MB f16 [B][N][64]
#define OFF_IDX8   (OFF_FTST  + 33554432ull)  // 8MB  i32 [B*N][8]
#define OFF_W8     (OFF_IDX8  + 8388608ull)   // 8MB  f32 [B*N][8]
#define OFF_PIDX   (OFF_W8    + 8388608ull)   // 1MB  i32 [B*N]
#define OFF_STATS  (OFF_PIDX  + 1048576ull)   // [B][4] mean.xyz, maxnorm
#define OFF_WFRAG1 (OFF_STATS + 256ull)       // 4*54*32*8 dwords
#define OFF_WFRAG2 (OFF_WFRAG1 + 221184ull)
#define OFF_PFRAG  (OFF_WFRAG2 + 221184ull)   // 4*2*32*8 dwords

// ---------------- per-batch coordinate stats ----------------
__global__ __launch_bounds__(256)
void coord_stats(const float* __restrict__ coords, float* __restrict__ stats)
{
    __shared__ float sr[256];
    int b = blockIdx.x;
    float m[3];
    for (int d = 0; d < 3; ++d) {
        const float* p = coords + ((b * 3 + d) << 15);
        float a = 0.f;
        for (int i = threadIdx.x; i < NPTS; i += 256) a += p[i];
        sr[threadIdx.x] = a; __syncthreads();
        for (int s = 128; s > 0; s >>= 1) {
            if (threadIdx.x < s) sr[threadIdx.x] += sr[threadIdx.x + s];
            __syncthreads();
        }
        m[d] = sr[0] * (1.f / 32768.f);
        __syncthreads();
    }
    const float* px = coords + ((b * 3 + 0) << 15);
    const float* py = coords + ((b * 3 + 1) << 15);
    const float* pz = coords + ((b * 3 + 2) << 15);
    float mx = 0.f;
    for (int i = threadIdx.x; i < NPTS; i += 256) {
        float dx = px[i] - m[0], dy = py[i] - m[1], dz = pz[i] - m[2];
        mx = fmaxf(mx, sqrtf(dx * dx + dy * dy + dz * dz));
    }
    sr[threadIdx.x] = mx; __syncthreads();
    for (int s = 128; s > 0; s >>= 1) {
        if (threadIdx.x < s) sr[threadIdx.x] = fmaxf(sr[threadIdx.x], sr[threadIdx.x + s]);
        __syncthreads();
    }
    if (threadIdx.x == 0) {
        stats[b * 4 + 0] = m[0]; stats[b * 4 + 1] = m[1];
        stats[b * 4 + 2] = m[2]; stats[b * 4 + 3] = sr[0];
    }
}

// ---------------- per-point voxel index + trilinear weights ----------------
__global__ __launch_bounds__(256)
void point_prep(const float* __restrict__ coords, const float* __restrict__ stats,
                int* __restrict__ pidx, int* __restrict__ idx8,
                float* __restrict__ w8, float* __restrict__ cnt)
{
    int t = blockIdx.x * 256 + threadIdx.x;   // [0, B*N)
    int b = t >> 15, n = t & (NPTS - 1);
    float inv = 0.5f / stats[b * 4 + 3];
    int lo[3], hi[3], vi[3];
    float fr[3];
#pragma unroll
    for (int d = 0; d < 3; ++d) {
        float c  = coords[((b * 3 + d) << 15) + n];
        float c01 = (c - stats[b * 4 + d]) * inv + 0.5f;
        float nc  = fminf(fmaxf(c01 * 32.f, 0.f), 31.f);
        vi[d] = (int)rintf(nc);               // round-half-even == jnp.round
        float fl = floorf(nc);
        int   l  = (int)fl;
        fr[d] = nc - fl;
        lo[d] = l;
        hi[d] = (l + 1 > 31) ? 31 : l + 1;
    }
    int vidx = vi[0] * 1024 + vi[1] * 32 + vi[2];
    pidx[t] = vidx;
    atomicAdd(&cnt[(b << 15) + vidx], 1.f);
#pragma unroll
    for (int k = 0; k < 8; ++k) {
        int dx = k >> 2, dy = (k >> 1) & 1, dz = k & 1;
        int ix = dx ? hi[0] : lo[0];
        int iy = dy ? hi[1] : lo[1];
        int iz = dz ? hi[2] : lo[2];
        float w = (dx ? fr[0] : 1.f - fr[0]) * (dy ? fr[1] : 1.f - fr[1]) *
                  (dz ? fr[2] : 1.f - fr[2]);
        idx8[t * 8 + k] = ix * 1024 + iy * 32 + iz;
        w8[t * 8 + k]   = w;
    }
}

// ---------------- scatter-add features into voxel sums ----------------
// one thread per point; loops 64 channels with the voxel index in a register
__global__ __launch_bounds__(256)
void scatter_feats(const float* __restrict__ feats, const int* __restrict__ pidx,
                   float* __restrict__ sums)
{
    int t = blockIdx.x * 256 + threadIdx.x;   // [0, B*N)
    int b = t >> 15, n = t & (NPTS - 1);
    int vi = pidx[t];
    float*       s = sums  + (b << 21) + vi;
    const float* f = feats + (b << 21) + n;
#pragma unroll 4
    for (int c = 0; c < 64; ++c)
        atomicAdd(s + (c << 15), f[c << 15]);
}

// ---------------- sums/count -> f16 grid ----------------
__global__ __launch_bounds__(256)
void finalize_grid(const float* __restrict__ sums, const float* __restrict__ cnt,
                   _Float16* __restrict__ g16)
{
    int t  = blockIdx.x * 256 + threadIdx.x;  // [0, B*64*32768)
    int v  = t & (RCUBE - 1);
    int bc = t >> 15;
    int b  = bc >> 6;
    float c = cnt[(b << 15) + v];
    g16[t] = (_Float16)(sums[t] / fmaxf(c, 1.f));
}

// ---------------- build WMMA A-fragments from f32 weights ----------------
// A-fragment k order used here (contiguous, B uses the same): lanes 0-15: K 0-15;
// lanes 16-31: K 16-31.  k_global = chunk*32 + klocal; for conv
// k = tap*64 + c with tap = kx*9+ky*3+kz, source w[o][c][kx][ky][kz].
__global__ __launch_bounds__(256)
void prep_afrag(const float* __restrict__ w, uint32_t* __restrict__ frag,
                int nchunks, int is_conv)
{
    int t = blockIdx.x * 256 + threadIdx.x;   // 4*nchunks*256
    int per   = nchunks << 8;
    int mt    = t / per;
    int rem   = t - mt * per;
    int chunk = rem >> 8;
    int lane  = (rem >> 3) & 31;
    int j     = rem & 7;
    int m     = mt * 16 + (lane & 15);
    int kl    = (lane & 16) + 2 * j;
    int k0 = chunk * 32 + kl;
    float w0, w1;
    if (is_conv) {
        int tap = k0 >> 6, c = k0 & 63;
        w0 = w[(m * 64 + c) * 27 + tap];
        w1 = w[(m * 64 + c + 1) * 27 + tap];
    } else {
        w0 = w[m * 64 + k0];
        w1 = w[m * 64 + k0 + 1];
    }
    union { _Float16 h[2]; uint32_t u; } p;
    p.h[0] = (_Float16)w0;
    p.h[1] = (_Float16)w1;
    frag[t] = p.u;
}

// ---------------- 3x3x3 conv as implicit GEMM via WMMA ----------------
// Workgroup = (b, x, y): output tile Cout(64) x z(32). 8 waves: mt 0..3, nt 0..1.
// K = 27 taps * 64 cin = 1728 -> 54 chunks of 32; chunk pairs share a tap.
// Dual accumulators break the WMMA->WMMA RAW chain.
__global__ __launch_bounds__(256)
void conv3d_wmma(const _Float16* __restrict__ in16, const uint32_t* __restrict__ afrag,
                 float* __restrict__ outp)
{
    __shared__ _Float16 Xs[9 * 34 * 64];      // [nbr(dx,dy)][z+halo][cin], 39168 B
    int blk = blockIdx.x;
    int b = blk >> 10, x = (blk >> 5) & 31, y = blk & 31;
    int tid = threadIdx.x;
    for (int e = tid; e < 9 * 34 * 64; e += 256) {
        int zz = e % 34;
        int rest = e / 34;
        int c = rest & 63, nbr = rest >> 6;
        int gx = x + nbr / 3 - 1, gy = y + nbr % 3 - 1, gz = zz - 1;
        _Float16 v = (_Float16)0.f;
        if ((unsigned)gx < 32u && (unsigned)gy < 32u && (unsigned)gz < 32u)
            v = in16[((b * 64 + c) << 15) + (gx << 10) + (gy << 5) + gz];
        Xs[(nbr * 34 + zz) * 64 + c] = v;
    }
    __syncthreads();
    const uint32_t* Xs32 = (const uint32_t*)Xs;
    int wv = tid >> 5, lane = tid & 31;
    int mt = wv >> 1, nt = wv & 1;
    int zloc = nt * 16 + (lane & 15);
    int chw  = (lane & 16) >> 1;              // +8 dwords for lanes 16-31 (B layout)
    v8f acc0 = {}, acc1 = {};
    for (int chunk = 0; chunk < 54; chunk += 2) {
        union { v16h h; uint32_t u[8]; } A0, A1, B0, B1;
        const uint32_t* ap = afrag + (((mt * 54 + chunk) * 32 + lane) << 3);
#pragma unroll
        for (int j = 0; j < 8; ++j) { A0.u[j] = ap[j]; A1.u[j] = ap[256 + j]; }
        __builtin_prefetch((const void*)(ap + 512), 0, 0);   // next chunk pair
        int tap = chunk >> 1;                 // chunk & chunk+1 share this tap
        int nbr = tap / 3, dz = tap % 3;
        int row = (nbr * 34 + zloc + dz) * 32;               // dword row
#pragma unroll
        for (int j = 0; j < 8; ++j) {
            B0.u[j] = Xs32[row + chw + j];
            B1.u[j] = Xs32[row + 16 + chw + j];
        }
        acc0 = __builtin_amdgcn_wmma_f32_16x16x32_f16(false, A0.h, false, B0.h,
                                                      (short)0, acc0, false, false);
        acc1 = __builtin_amdgcn_wmma_f32_16x16x32_f16(false, A1.h, false, B1.h,
                                                      (short)0, acc1, false, false);
    }
    v8f acc = acc0 + acc1;
    int ob   = mt * 16 + ((lane >> 4) << 3);
    int base = ((b * 64 + ob) << 15) + (x << 10) + (y << 5) + zloc;
#pragma unroll
    for (int v = 0; v < 8; ++v)
        outp[base + (v << 15)] = acc[v];
}

// ---------------- instance norm + leaky relu (per b,channel over 32768) ----------------
__global__ __launch_bounds__(256)
void inorm_leaky(const float* __restrict__ in, float* __restrict__ outf,
                 _Float16* __restrict__ outh)
{
    __shared__ float s1[256], s2[256];
    int ch = blockIdx.x;                      // b*64 + c
    const float* p = in + (ch << 15);
    float a = 0.f, q = 0.f;
    for (int i = threadIdx.x; i < RCUBE; i += 256) {
        float v = p[i];
        a += v; q += v * v;
    }
    s1[threadIdx.x] = a; s2[threadIdx.x] = q;
    __syncthreads();
    for (int s = 128; s > 0; s >>= 1) {
        if (threadIdx.x < s) {
            s1[threadIdx.x] += s1[threadIdx.x + s];
            s2[threadIdx.x] += s2[threadIdx.x + s];
        }
        __syncthreads();
    }
    float mean = s1[0] * (1.f / 32768.f);
    float var  = s2[0] * (1.f / 32768.f) - mean * mean;
    float rs   = rsqrtf(var + 1e-4f);
    for (int i = threadIdx.x; i < RCUBE; i += 256) {
        float v = (p[i] - mean) * rs;
        v = (v >= 0.f) ? v : 0.1f * v;
        if (outf) outf[(ch << 15) + i] = v;
        if (outh) outh[(ch << 15) + i] = (_Float16)v;
    }
}

// ---------------- features [B][C][N] f32 -> [B][N][C] f16 (LDS tiled) ----------------
__global__ __launch_bounds__(256)
void cvt_ftsT(const float* __restrict__ f, _Float16* __restrict__ o)
{
    __shared__ _Float16 tile[64][65];
    int blk = blockIdx.x;                     // B * (N/64)
    int b = blk >> 9;
    int n0 = (blk & 511) << 6;
    int tid = threadIdx.x;
    for (int e = tid; e < 4096; e += 256) {
        int i = e & 63, c = e >> 6;
        tile[c][i] = (_Float16)f[((b * 64 + c) << 15) + n0 + i];
    }
    __syncthreads();
    for (int e = tid; e < 4096; e += 256) {
        int c = e & 63, i = e >> 6;
        o[(((b << 15) + n0 + i) << 6) + c] = tile[c][i];
    }
}

// ---------------- pointwise MLP GEMM via WMMA: p = pw[64x64] @ ftsT ----------------
__global__ __launch_bounds__(256)
void pgemm_wmma(const _Float16* __restrict__ ftsT, const uint32_t* __restrict__ afrag,
                float* __restrict__ outp)
{
    int blk = blockIdx.x;                     // B * (N/32)
    int b  = blk >> 10;
    int p0 = (blk & 1023) << 5;
    int tid = threadIdx.x, wv = tid >> 5, lane = tid & 31;
    int mt = wv >> 1, nt = wv & 1;
    int pt  = p0 + nt * 16 + (lane & 15);
    int chw = (lane & 16) >> 1;
    const uint32_t* bbase = (const uint32_t*)(ftsT + (((b << 15) + pt) << 6));
    union { v16h h; uint32_t u[8]; } A0, A1, B0, B1;
    const uint32_t* ap = afrag + (((mt * 2) * 32 + lane) << 3);
#pragma unroll
    for (int j = 0; j < 8; ++j) { A0.u[j] = ap[j]; A1.u[j] = ap[256 + j]; }
#pragma unroll
    for (int j = 0; j < 8; ++j) {
        B0.u[j] = bbase[chw + j];
        B1.u[j] = bbase[16 + chw + j];
    }
    v8f acc0 = {}, acc1 = {};
    acc0 = __builtin_amdgcn_wmma_f32_16x16x32_f16(false, A0.h, false, B0.h,
                                                  (short)0, acc0, false, false);
    acc1 = __builtin_amdgcn_wmma_f32_16x16x32_f16(false, A1.h, false, B1.h,
                                                  (short)0, acc1, false, false);
    v8f acc = acc0 + acc1;
    int ob   = mt * 16 + ((lane >> 4) << 3);
    int base = ((b * 64 + ob) << 15) + pt;
#pragma unroll
    for (int v = 0; v < 8; ++v)
        outp[base + (v << 15)] = acc[v];
}

// ---------------- trilinear gather + fuse ----------------
// one thread per point; 8 idx/weights held in registers across all 64 channels
__global__ __launch_bounds__(256)
void devox_fuse(const float* __restrict__ p, const float* __restrict__ vox,
                const int* __restrict__ idx8, const float* __restrict__ w8,
                float* __restrict__ fused)
{
    int t = blockIdx.x * 256 + threadIdx.x;   // [0, B*N)
    int b = t >> 15, n = t & (NPTS - 1);
    int   id[8];
    float wt[8];
#pragma unroll
    for (int k = 0; k < 8; ++k) { id[k] = idx8[t * 8 + k]; wt[k] = w8[t * 8 + k]; }
    const float* pb = p     + (b << 21) + n;
    const float* vb = vox   + (b << 21);
    float*       fb = fused + (b << 21) + n;
#pragma unroll 2
    for (int c = 0; c < 64; ++c) {
        float acc = pb[c << 15];
        const float* v = vb + (c << 15);
#pragma unroll
        for (int k = 0; k < 8; ++k)
            acc += v[id[k]] * wt[k];
        fb[c << 15] = acc;
    }
}

extern "C" void kernel_launch(void* const* d_in, const int* in_sizes, int n_in,
                              void* d_out, int out_size, void* d_ws, size_t ws_size,
                              hipStream_t stream)
{
    (void)in_sizes; (void)n_in; (void)out_size; (void)ws_size;
    const float* features = (const float*)d_in[0];
    const float* coords   = (const float*)d_in[1];
    const float* c1w      = (const float*)d_in[2];
    const float* c2w      = (const float*)d_in[4];
    const float* pw       = (const float*)d_in[6];
    // conv biases / pb cancel exactly under InstanceNorm -> skipped.

    char* ws = (char*)d_ws;
    float*     sums   = (float*)(ws + OFF_SUMS);
    float*     cnt    = (float*)(ws + OFF_CNT);
    _Float16*  g16    = (_Float16*)(ws + OFF_G16);   // grid f16, later h f16
    _Float16*  ftsT   = (_Float16*)(ws + OFF_FTST);
    int*       idx8   = (int*)(ws + OFF_IDX8);
    float*     w8     = (float*)(ws + OFF_W8);
    int*       pidx   = (int*)(ws + OFF_PIDX);
    float*     stats  = (float*)(ws + OFF_STATS);
    uint32_t*  wfrag1 = (uint32_t*)(ws + OFF_WFRAG1);
    uint32_t*  wfrag2 = (uint32_t*)(ws + OFF_WFRAG2);
    uint32_t*  pfrag  = (uint32_t*)(ws + OFF_PFRAG);

    float* out        = (float*)d_out;
    float* out_fused  = out;                     // [8,64,32768]
    float* out_coords = out + 16777216;          // [8,3,32768]
    float* out_vox    = out + 16777216 + 786432; // [8,64,32,32,32]

    hipMemsetAsync(sums, 0, 67108864ull, stream);
    hipMemsetAsync(cnt,  0, 1048576ull,  stream);

    coord_stats  <<<8,    256, 0, stream>>>(coords, stats);
    point_prep   <<<1024, 256, 0, stream>>>(coords, stats, pidx, idx8, w8, cnt);
    scatter_feats<<<1024, 256, 0, stream>>>(features, pidx, sums);
    finalize_grid<<<65536,256, 0, stream>>>(sums, cnt, g16);

    prep_afrag<<<216, 256, 0, stream>>>(c1w, wfrag1, 54, 1);
    prep_afrag<<<216, 256, 0, stream>>>(c2w, wfrag2, 54, 1);
    prep_afrag<<<8,   256, 0, stream>>>(pw,  pfrag,  2,  0);

    // voxel branch
    conv3d_wmma<<<8192, 256, 0, stream>>>(g16, wfrag1, sums);       // conv1 -> sums
    inorm_leaky<<<512,  256, 0, stream>>>(sums, nullptr, g16);      // -> h f16 (reuse)
    conv3d_wmma<<<8192, 256, 0, stream>>>(g16, wfrag2, sums);       // conv2 -> sums
    inorm_leaky<<<512,  256, 0, stream>>>(sums, out_vox, nullptr);  // voxel_features

    // point branch
    cvt_ftsT  <<<4096, 256, 0, stream>>>(features, ftsT);
    pgemm_wmma<<<8192, 256, 0, stream>>>(ftsT, pfrag, sums);        // p -> sums (reuse)
    inorm_leaky<<<512, 256, 0, stream>>>(sums, sums, nullptr);      // in-place norm

    // devoxelize + fuse
    devox_fuse<<<1024, 256, 0, stream>>>(sums, out_vox, idx8, w8, out_fused);

    // coords passthrough
    hipMemcpyAsync(out_coords, coords, 786432ull * 4ull,
                   hipMemcpyDeviceToDevice, stream);
}
